// AttentionBlock_56573309223252
// MI455X (gfx1250) — matrix operand order repaired and checked by reading, not compiled, not run on hardware
//
#include <hip/hip_runtime.h>
#include <stdint.h>

#define B_   16
#define C_   512
#define C3_  1536
#define N_   1024
#define G_   32
#define CPG_ 16
#define EPS_ 1e-5f

typedef __attribute__((ext_vector_type(16))) __bf16 v16bf;
typedef __attribute__((ext_vector_type(8)))  float  v8f;

union FragU { unsigned int u[8]; v16bf v; };
union Pack8 { unsigned short s[8]; uint4 u4; };

__device__ __forceinline__ unsigned short f2bf(float x) {
  unsigned int u = __float_as_uint(x);
  u += 0x7FFFu + ((u >> 16) & 1u);
  return (unsigned short)(u >> 16);
}

__device__ __forceinline__ v8f wmma_bf16(v16bf a, v16bf b, v8f c) {
  return __builtin_amdgcn_wmma_f32_16x16x32_bf16(false, a, false, b, (short)0, c, false, false);
}

// A fragment (16x32 bf16): lane row = lane&15; lanes0-15 K=0..7,16..23; lanes16-31 +8.
// Memory layout: element (row, K) at p[row*pitch + K] (K contiguous) -> two b128 loads.
__device__ __forceinline__ v16bf load_frag_A(const unsigned short* p, int row, int pitch,
                                             int k0, int lane) {
  FragU f;
  const int hi = (lane >> 4) & 1;
  const unsigned short* rp = p + (size_t)row * pitch + k0 + hi * 8;
#pragma unroll
  for (int j = 0; j < 4; ++j) f.u[j] = *(const unsigned int*)(rp + 2 * j);
#pragma unroll
  for (int j = 0; j < 4; ++j) f.u[4 + j] = *(const unsigned int*)(rp + 16 + 2 * j);
  return f.v;
}

// B fragment (32x16 bf16): lane col = lane&15; lanes0-15 K=0..7,16..23; lanes16-31 +8.
// Memory layout: element (K, col) at p[col*pitch + K] (K contiguous) -> two b128 loads.
__device__ __forceinline__ v16bf load_frag_B(const unsigned short* p, int col, int pitch,
                                             int k0, int lane) {
  FragU f;
  const int hi = (lane >> 4) & 1;
  const unsigned short* cp = p + (size_t)col * pitch + k0 + hi * 8;
#pragma unroll
  for (int j = 0; j < 4; ++j) f.u[j] = *(const unsigned int*)(cp + 2 * j);
#pragma unroll
  for (int j = 0; j < 4; ++j) f.u[4 + j] = *(const unsigned int*)(cp + 16 + 2 * j);
  return f.v;
}

// ---------------- precision convert ----------------
__global__ void k_f32_to_bf16(const float* __restrict__ src, unsigned short* __restrict__ dst, int n) {
  int i = blockIdx.x * 256 + threadIdx.x;
  if (i < n) dst[i] = f2bf(src[i]);
}

// ---------------- group norm ----------------
__global__ void k_gn_stats(const float* __restrict__ x, float* __restrict__ stats) {
  __shared__ float s1[256], s2[256];
  const int bg = blockIdx.x;                            // b*G + g
  const float* base = x + (size_t)bg * (CPG_ * N_);     // group channels contiguous
  float s = 0.f, ss = 0.f;
  for (int i = threadIdx.x; i < CPG_ * N_; i += 256) {
    float v = base[i];
    s += v; ss += v * v;
  }
  const int t = threadIdx.x;
  s1[t] = s; s2[t] = ss;
  __syncthreads();
  for (int w = 128; w > 0; w >>= 1) {
    if (t < w) { s1[t] += s1[t + w]; s2[t] += s2[t + w]; }
    __syncthreads();
  }
  if (t == 0) {
    float inv = 1.0f / (float)(CPG_ * N_);
    float mean = s1[0] * inv;
    float var = s2[0] * inv - mean * mean;
    stats[2 * bg] = mean;
    stats[2 * bg + 1] = rsqrtf(var + EPS_);
  }
}

// writes h in token-major [B, N, C] bf16
__global__ void k_gn_apply(const float* __restrict__ x, const float* __restrict__ stats,
                           const float* __restrict__ gw, const float* __restrict__ gb,
                           unsigned short* __restrict__ h) {
  int i4 = (blockIdx.x * 256 + threadIdx.x) * 4;
  int b = i4 / (C_ * N_);
  int rem = i4 % (C_ * N_);
  int c = rem / N_;
  int n = rem % N_;                                     // multiple of 4
  int g = c >> 4;
  float mean = stats[2 * (b * G_ + g)];
  float rstd = stats[2 * (b * G_ + g) + 1];
  float w = gw[c] * rstd;
  float bb = gb[c];
  float4 xv = *(const float4*)(x + i4);
  size_t base = (size_t)(b * N_ + n) * C_ + c;
  h[base]          = f2bf((xv.x - mean) * w + bb);
  h[base + C_]     = f2bf((xv.y - mean) * w + bb);
  h[base + 2 * C_] = f2bf((xv.z - mean) * w + bb);
  h[base + 3 * C_] = f2bf((xv.w - mean) * w + bb);
}

// ---------------- GEMM1: qkv = W_qkv @ h ; 32x32 macro-tile per wave (2x2 WMMA) ----------------
__global__ void k_gemm_qkv(const unsigned short* __restrict__ wq, const float* __restrict__ qkv_b,
                           const unsigned short* __restrict__ h,
                           unsigned short* __restrict__ q, unsigned short* __restrict__ k,
                           unsigned short* __restrict__ v) {
  const int lane = threadIdx.x & 31;
  const int tile = blockIdx.x * 8 + (threadIdx.x >> 5); // macro tiles: 48 x 32 per batch
  const int b = tile / (48 * 32);
  const int rr = tile % (48 * 32);
  const int MT = rr / 32, NT = rr % 32;
  const unsigned short* hb = h + (size_t)b * N_ * C_;
  const int l15 = lane & 15;
  const int ar0 = MT * 32 + l15, ar1 = ar0 + 16;
  const int bc0 = NT * 32 + l15, bc1 = bc0 + 16;
  v8f acc00 = {}, acc01 = {}, acc10 = {}, acc11 = {};
  for (int kk = 0; kk < C_; kk += 32) {
    v16bf a0 = load_frag_A(wq, ar0, C_, kk, lane);
    v16bf a1 = load_frag_A(wq, ar1, C_, kk, lane);
    v16bf b0 = load_frag_B(hb, bc0, C_, kk, lane);
    v16bf b1 = load_frag_B(hb, bc1, C_, kk, lane);
    acc00 = wmma_bf16(a0, b0, acc00);
    acc01 = wmma_bf16(a0, b1, acc01);
    acc10 = wmma_bf16(a1, b0, acc10);
    acc11 = wmma_bf16(a1, b1, acc11);
  }
  const int hi = (lane >> 4) & 1;
  v8f accs[2][2] = {{acc00, acc01}, {acc10, acc11}};
#pragma unroll
  for (int i = 0; i < 2; ++i) {
    const int obase = (MT * 2 + i) * 16 + hi * 8;       // 8 consecutive output channels
    float4 b0v = *(const float4*)(qkv_b + obase);
    float4 b1v = *(const float4*)(qkv_b + obase + 4);
    float bias[8] = {b0v.x, b0v.y, b0v.z, b0v.w, b1v.x, b1v.y, b1v.z, b1v.w};
#pragma unroll
    for (int j = 0; j < 2; ++j) {
      const int ncol = (NT * 2 + j) * 16 + l15;
      if (MT < 16) {                                    // Q -> [B,N,C]
        Pack8 pk;
#pragma unroll
        for (int r = 0; r < 8; ++r) pk.s[r] = f2bf(accs[i][j][r] + bias[r]);
        *(uint4*)(q + (size_t)(b * N_ + ncol) * C_ + obase) = pk.u4;
      } else if (MT < 32) {                             // K -> [B,N,C]
        Pack8 pk;
#pragma unroll
        for (int r = 0; r < 8; ++r) pk.s[r] = f2bf(accs[i][j][r] + bias[r]);
        *(uint4*)(k + (size_t)(b * N_ + ncol) * C_ + (obase - 512)) = pk.u4;
      } else {                                          // V -> [B,C,N]
#pragma unroll
        for (int r = 0; r < 8; ++r) {
          int c = obase - 1024 + r;
          v[(size_t)(b * C_ + c) * N_ + ncol] = f2bf(accs[i][j][r] + bias[r]);
        }
      }
    }
  }
}

// ---------------- GEMM2: S = Q @ K^T (f32) ; 32x32 macro-tile ----------------
__global__ void k_gemm_s(const unsigned short* __restrict__ q, const unsigned short* __restrict__ kt,
                         float* __restrict__ S) {
  const int lane = threadIdx.x & 31;
  const int tile = blockIdx.x * 8 + (threadIdx.x >> 5); // 32 x 32 macro tiles per batch
  const int b = tile / (32 * 32);
  const int rr = tile % (32 * 32);
  const int MT = rr / 32, NT = rr % 32;
  const unsigned short* qb = q + (size_t)b * N_ * C_;
  const unsigned short* kb = kt + (size_t)b * N_ * C_;
  const int l15 = lane & 15;
  const int ar0 = MT * 32 + l15, ar1 = ar0 + 16;
  const int bc0 = NT * 32 + l15, bc1 = bc0 + 16;
  v8f acc00 = {}, acc01 = {}, acc10 = {}, acc11 = {};
  for (int kk = 0; kk < C_; kk += 32) {
    v16bf a0 = load_frag_A(qb, ar0, C_, kk, lane);
    v16bf a1 = load_frag_A(qb, ar1, C_, kk, lane);
    v16bf b0 = load_frag_B(kb, bc0, C_, kk, lane);
    v16bf b1 = load_frag_B(kb, bc1, C_, kk, lane);
    acc00 = wmma_bf16(a0, b0, acc00);
    acc01 = wmma_bf16(a0, b1, acc01);
    acc10 = wmma_bf16(a1, b0, acc10);
    acc11 = wmma_bf16(a1, b1, acc11);
  }
  const int hi = (lane >> 4) & 1;
  float* Sb = S + (size_t)b * N_ * N_;
  v8f accs[2][2] = {{acc00, acc01}, {acc10, acc11}};
#pragma unroll
  for (int i = 0; i < 2; ++i) {
#pragma unroll
    for (int j = 0; j < 2; ++j) {
      const int m = (NT * 2 + j) * 16 + l15;
#pragma unroll
      for (int r = 0; r < 8; ++r) {
        int n = (MT * 2 + i) * 16 + r + hi * 8;
        Sb[(size_t)n * N_ + m] = accs[i][j][r];
      }
    }
  }
}

// ---------------- row softmax; write P bf16 in place over S rows (pitch 2N halves) ----------------
__global__ void k_softmax(float* __restrict__ S) {
  __shared__ float red[256];
  float* row = S + (size_t)blockIdx.x * N_;
  const int t = threadIdx.x;
  const float scale = 0.04419417382415922f;             // 1/sqrt(512)
  float4 xv = *(const float4*)(row + t * 4);
  float v0 = xv.x * scale, v1 = xv.y * scale, v2 = xv.z * scale, v3 = xv.w * scale;
  float mx = fmaxf(fmaxf(v0, v1), fmaxf(v2, v3));
  red[t] = mx;
  __syncthreads();
  for (int w = 128; w > 0; w >>= 1) { if (t < w) red[t] = fmaxf(red[t], red[t + w]); __syncthreads(); }
  float rmax = red[0];
  __syncthreads();
  float e0 = __expf(v0 - rmax), e1 = __expf(v1 - rmax);
  float e2 = __expf(v2 - rmax), e3 = __expf(v3 - rmax);
  red[t] = e0 + e1 + e2 + e3;
  __syncthreads();
  for (int w = 128; w > 0; w >>= 1) { if (t < w) red[t] += red[t + w]; __syncthreads(); }
  float inv = 1.0f / red[0];
  unsigned short* prow = (unsigned short*)row;          // all row reads precede first barrier
  union { unsigned short s[4]; uint2 u2; } pk;
  pk.s[0] = f2bf(e0 * inv); pk.s[1] = f2bf(e1 * inv);
  pk.s[2] = f2bf(e2 * inv); pk.s[3] = f2bf(e3 * inv);
  *(uint2*)(prow + t * 4) = pk.u2;
}

// ---------------- GEMM3: O = P @ V (O -> [B,N,C] bf16, reuses h) ; 32x32 macro-tile ----------------
__global__ void k_gemm_pv(const unsigned short* __restrict__ P, const unsigned short* __restrict__ v,
                          unsigned short* __restrict__ O) {
  const int lane = threadIdx.x & 31;
  const int tile = blockIdx.x * 8 + (threadIdx.x >> 5); // 32 x 16 macro tiles per batch
  const int b = tile / (32 * 16);
  const int rr = tile % (32 * 16);
  const int MT = rr / 16, NT = rr % 16;
  const unsigned short* Pb = P + (size_t)b * N_ * (2 * N_);  // bf16 rows, pitch 2N
  const unsigned short* vb = v + (size_t)b * C_ * N_;
  const int l15 = lane & 15;
  const int ar0 = MT * 32 + l15, ar1 = ar0 + 16;
  const int bc0 = NT * 32 + l15, bc1 = bc0 + 16;
  v8f acc00 = {}, acc01 = {}, acc10 = {}, acc11 = {};
  for (int kk = 0; kk < N_; kk += 32) {
    v16bf a0 = load_frag_A(Pb, ar0, 2 * N_, kk, lane);
    v16bf a1 = load_frag_A(Pb, ar1, 2 * N_, kk, lane);
    v16bf b0 = load_frag_B(vb, bc0, N_, kk, lane);
    v16bf b1 = load_frag_B(vb, bc1, N_, kk, lane);
    acc00 = wmma_bf16(a0, b0, acc00);
    acc01 = wmma_bf16(a0, b1, acc01);
    acc10 = wmma_bf16(a1, b0, acc10);
    acc11 = wmma_bf16(a1, b1, acc11);
  }
  const int hi = (lane >> 4) & 1;
  v8f accs[2][2] = {{acc00, acc01}, {acc10, acc11}};
#pragma unroll
  for (int i = 0; i < 2; ++i) {
#pragma unroll
    for (int j = 0; j < 2; ++j) {
      const int c = (NT * 2 + j) * 16 + l15;
#pragma unroll
      for (int r = 0; r < 8; ++r) {
        int n = (MT * 2 + i) * 16 + r + hi * 8;
        O[(size_t)(b * N_ + n) * C_ + c] = f2bf(accs[i][j][r]);
      }
    }
  }
}

// ---------------- GEMM4: out = proj_w @ O + proj_b + x ; 32x32 macro-tile ----------------
__global__ void k_gemm_proj(const unsigned short* __restrict__ O, const unsigned short* __restrict__ wp,
                            const float* __restrict__ proj_b, const float* __restrict__ x,
                            float* __restrict__ out) {
  const int lane = threadIdx.x & 31;
  const int tile = blockIdx.x * 8 + (threadIdx.x >> 5); // 32 x 16 macro tiles per batch
  const int b = tile / (32 * 16);
  const int rr = tile % (32 * 16);
  const int MT = rr / 16, NT = rr % 16;                 // MT: n rows, NT: o cols
  const unsigned short* Ob = O + (size_t)b * N_ * C_;
  const int l15 = lane & 15;
  const int ar0 = MT * 32 + l15, ar1 = ar0 + 16;
  const int bc0 = NT * 32 + l15, bc1 = bc0 + 16;
  v8f acc00 = {}, acc01 = {}, acc10 = {}, acc11 = {};
  for (int kk = 0; kk < C_; kk += 32) {
    v16bf a0 = load_frag_A(Ob, ar0, C_, kk, lane);
    v16bf a1 = load_frag_A(Ob, ar1, C_, kk, lane);
    v16bf b0 = load_frag_B(wp, bc0, C_, kk, lane);
    v16bf b1 = load_frag_B(wp, bc1, C_, kk, lane);
    acc00 = wmma_bf16(a0, b0, acc00);
    acc01 = wmma_bf16(a0, b1, acc01);
    acc10 = wmma_bf16(a1, b0, acc10);
    acc11 = wmma_bf16(a1, b1, acc11);
  }
  const int hi = (lane >> 4) & 1;
  v8f accs[2][2] = {{acc00, acc01}, {acc10, acc11}};
#pragma unroll
  for (int j = 0; j < 2; ++j) {
    const int o = (NT * 2 + j) * 16 + l15;
    const float pb = proj_b[o];
#pragma unroll
    for (int i = 0; i < 2; ++i) {
      const int nbase = (MT * 2 + i) * 16 + hi * 8;     // 8 consecutive n
      const size_t off = (size_t)(b * C_ + o) * N_ + nbase;
      float4 x0 = *(const float4*)(x + off);
      float4 x1 = *(const float4*)(x + off + 4);
      float4 r0, r1;
      r0.x = accs[i][j][0] + pb + x0.x; r0.y = accs[i][j][1] + pb + x0.y;
      r0.z = accs[i][j][2] + pb + x0.z; r0.w = accs[i][j][3] + pb + x0.w;
      r1.x = accs[i][j][4] + pb + x1.x; r1.y = accs[i][j][5] + pb + x1.y;
      r1.z = accs[i][j][6] + pb + x1.z; r1.w = accs[i][j][7] + pb + x1.w;
      *(float4*)(out + off) = r0;
      *(float4*)(out + off + 4) = r1;
    }
  }
}

extern "C" void kernel_launch(void* const* d_in, const int* in_sizes, int n_in,
                              void* d_out, int out_size, void* d_ws, size_t ws_size,
                              hipStream_t stream) {
  const float* x      = (const float*)d_in[0];
  const float* gn_w   = (const float*)d_in[1];
  const float* gn_b   = (const float*)d_in[2];
  const float* qkv_w  = (const float*)d_in[3];
  const float* qkv_b  = (const float*)d_in[4];
  const float* proj_w = (const float*)d_in[5];
  const float* proj_b = (const float*)d_in[6];
  float* out = (float*)d_out;
  char* ws = (char*)d_ws;

  // workspace layout (bytes)
  unsigned short* wq    = (unsigned short*)(ws + 0);          // 1536*512*2  = 1572864
  unsigned short* wp    = (unsigned short*)(ws + 1572864);    // 512*512*2   = 524288
  float*          stats = (float*)         (ws + 2097152);    // 512*2*4     = 4096
  unsigned short* h     = (unsigned short*)(ws + 2101248);    // [B,N,C] bf16 = 16777216 (reused as O)
  unsigned short* q     = (unsigned short*)(ws + 18878464);   // [B,N,C] bf16
  unsigned short* k     = (unsigned short*)(ws + 35655680);   // [B,N,C] bf16
  unsigned short* v     = (unsigned short*)(ws + 52432896);   // [B,C,N] bf16
  float*          S     = (float*)         (ws + 69210112);   // [B,N,N] f32 = 67108864 (P bf16 in place)

  k_f32_to_bf16<<<(C3_ * C_ + 255) / 256, 256, 0, stream>>>(qkv_w, wq, C3_ * C_);
  k_f32_to_bf16<<<(C_ * C_ + 255) / 256, 256, 0, stream>>>(proj_w, wp, C_ * C_);
  k_gn_stats<<<B_ * G_, 256, 0, stream>>>(x, stats);
  k_gn_apply<<<(B_ * C_ * N_) / (256 * 4), 256, 0, stream>>>(x, stats, gn_w, gn_b, h);
  k_gemm_qkv<<<(B_ * 48 * 32) / 8, 256, 0, stream>>>(wq, qkv_b, h, q, k, v);
  k_gemm_s<<<(B_ * 32 * 32) / 8, 256, 0, stream>>>(q, k, S);
  k_softmax<<<B_ * N_, 256, 0, stream>>>(S);
  k_gemm_pv<<<(B_ * 32 * 16) / 8, 256, 0, stream>>>((const unsigned short*)S, v, h);  // O reuses h
  k_gemm_proj<<<(B_ * 32 * 16) / 8, 256, 0, stream>>>(h, wp, proj_b, x, out);
}
